// Qformer_19215683682330
// MI455X (gfx1250) — compile-verified, matrix operand order checked by reading, not run on hardware
//
#include <hip/hip_runtime.h>
#include <hip/hip_bf16.h>

typedef __attribute__((ext_vector_type(16))) __bf16 v16bf;
typedef __attribute__((ext_vector_type(8)))  __bf16 v8bf;
typedef __attribute__((ext_vector_type(4)))  __bf16 v4bf;
typedef __attribute__((ext_vector_type(2)))  __bf16 v2bf;
typedef __attribute__((ext_vector_type(8)))  float  v8f;

#define SDIM 8
#define BDIM 128
#define NDIM 32
#define DMODEL 1024
#define INNER 1024
#define HEADS 16
#define DH 64
#define QN 64
#define WSTRIDE 40          // padded [n][k] stage stride (80 B: 16B-aligned, conflict-reducing)
#define ATT_SCALE 0.125f
#define LN_EPS 1e-5f

// ---------------- helpers ----------------
__device__ __forceinline__ v8f zero8() {
    v8f z;
#pragma unroll
    for (int i = 0; i < 8; ++i) z[i] = 0.0f;
    return z;
}

// WMMA 16x32 bf16 fragment from LDS tile stored [idx][k] row-major, contiguous K.
// ISA 7.12.2: VGPR0-3 hold K base..base+7, VGPR4-7 hold K base+16..base+23 (lanes>=16: +8)
// => exactly two ds_load_b128 per fragment.
__device__ __forceinline__ v16bf frag_ld(const __bf16* base, int idx0, int stride, int k0, int lane) {
    const int l16 = lane & 15, lhi = lane >> 4;
    const __bf16* p = base + (idx0 + l16) * stride + k0 + lhi * 8;
    v8bf lo = *(const v8bf*)p;          // K: base .. base+7
    v8bf hi = *(const v8bf*)(p + 16);   // K: base+16 .. base+23
    return __builtin_shufflevector(lo, hi, 0,1,2,3,4,5,6,7,8,9,10,11,12,13,14,15);
}

__device__ __forceinline__ v8f wmma_bf16(v16bf a, v16bf b, v8f c) {
    return __builtin_amdgcn_wmma_f32_16x16x32_bf16(false, a, false, b, (short)0, c, false, false);
}

__device__ __forceinline__ void store_bf4(__bf16* dst, float a, float b, float c, float d) {
    v4bf t; t[0] = (__bf16)a; t[1] = (__bf16)b; t[2] = (__bf16)c; t[3] = (__bf16)d;
    *(v4bf*)dst = t;   // ds_store_b64
}
__device__ __forceinline__ void store_bf2(__bf16* dst, float a, float b) {
    v2bf t; t[0] = (__bf16)a; t[1] = (__bf16)b;
    *(v2bf*)dst = t;   // ds_store_b32
}

// =====================================================================================
// Kernel 1: fused KV projection + multi-head cross attention. One WG per (s,n).
// X_sn staged once in LDS as bf16 -> x crosses HBM exactly once (134 MB).
// W_kv / q stream from L2 (8.25 MB resident).
// =====================================================================================
__global__ __launch_bounds__(256)
void attn_kernel(const float* __restrict__ x, const int* __restrict__ mask,
                 const float* __restrict__ q, const float* __restrict__ wkv,
                 float* __restrict__ attn_out)
{
    __shared__ __align__(16) __bf16 Xs[BDIM * DMODEL];   // 256 KB  [j][d]
    __shared__ __align__(16) __bf16 Wt[DH * WSTRIDE];    // 5 KB    W_kv slice, transposed [n][k]
    __shared__ __align__(16) __bf16 KVb[DH * BDIM];      // 16 KB   K as [j][64]  /  V^T as [d][128]
    __shared__ __align__(16) __bf16 Qb[QN * DH];         // 8 KB    [q][d]
    __shared__ __align__(16) float  Sim[QN * BDIM];      // 32 KB   logits; first 16KB aliased as bf16 attn
    __bf16* AttnB = (__bf16*)Sim;

    const int wg = blockIdx.x;
    const int s = wg >> 5, n = wg & 31;
    const int tid = threadIdx.x, lane = tid & 31, wv = tid >> 5;
    const int lrow = lane & 15, lhi = lane >> 4;

    // ---- stage X_sn: float4 global loads, packed bf16x4 LDS stores ----
    for (int it = tid; it < BDIM * (DMODEL / 4); it += 256) {
        int b = it >> 8, d4 = it & 255;
        float4 v = ((const float4*)(x + ((size_t)(s * BDIM + b) * NDIM + n) * DMODEL))[d4];
        store_bf4(&Xs[b * DMODEL + d4 * 4], v.x, v.y, v.z, v.w);
    }
    __syncthreads();

    for (int h = 0; h < HEADS; ++h) {
        // ---- stage q head slice [q][d] ----
        for (int it = tid; it < QN * (DH / 4); it += 256) {
            int r = it >> 4, c4 = (it & 15) * 4;
            float4 v = *(const float4*)(q + (size_t)r * INNER + h * DH + c4);
            store_bf4(&Qb[r * DH + c4], v.x, v.y, v.z, v.w);
        }

        // ---- K_h = X(128x1024) @ W_kv[:, h*64..]; wave wv owns j-rows 16wv.. ----
        v8f accK[4];
#pragma unroll
        for (int nt = 0; nt < 4; ++nt) accK[nt] = zero8();
        for (int kk = 0; kk < 32; ++kk) {
            __syncthreads();
            const size_t wbase = (size_t)(kk * 32) * (2 * INNER) + h * DH;
            if (kk + 1 < 32) __builtin_prefetch(wkv + wbase + 32 * (2 * INNER), 0, 3);
            for (int it = tid; it < 512; it += 256) {            // 32k x 64n, transposed stage
                int k = it >> 4, n4 = (it & 15) * 4;
                float4 v = *(const float4*)(wkv + wbase + (size_t)k * (2 * INNER) + n4);
                Wt[(n4 + 0) * WSTRIDE + k] = (__bf16)v.x;
                Wt[(n4 + 1) * WSTRIDE + k] = (__bf16)v.y;
                Wt[(n4 + 2) * WSTRIDE + k] = (__bf16)v.z;
                Wt[(n4 + 3) * WSTRIDE + k] = (__bf16)v.w;
            }
            __syncthreads();
            v16bf a = frag_ld(Xs, wv * 16, DMODEL, kk * 32, lane);
#pragma unroll
            for (int nt = 0; nt < 4; ++nt)
                accK[nt] = wmma_bf16(a, frag_ld(Wt, nt * 16, WSTRIDE, 0, lane), accK[nt]);
        }
        __syncthreads();
#pragma unroll
        for (int nt = 0; nt < 4; ++nt)
#pragma unroll
            for (int r = 0; r < 8; ++r)
                KVb[(wv * 16 + r + 8 * lhi) * DH + nt * 16 + lrow] = (__bf16)accK[nt][r];
        __syncthreads();

        // ---- sim = qh(64x64) @ K^T; wave wv owns j-cols 16wv.. ----
        v8f accS[4];
#pragma unroll
        for (int mt = 0; mt < 4; ++mt) accS[mt] = zero8();
#pragma unroll
        for (int ks = 0; ks < 2; ++ks) {
            v16bf b = frag_ld(KVb, wv * 16, DH, ks * 32, lane);  // [n=j][k=d]
#pragma unroll
            for (int mt = 0; mt < 4; ++mt)
                accS[mt] = wmma_bf16(frag_ld(Qb, mt * 16, DH, ks * 32, lane), b, accS[mt]);
        }
        {
            const int j  = wv * 16 + lrow;
            const int mv = mask[s * BDIM + j];
#pragma unroll
            for (int mt = 0; mt < 4; ++mt)
#pragma unroll
                for (int r = 0; r < 8; ++r) {
                    float v = accS[mt][r] * ATT_SCALE;
                    Sim[(mt * 16 + r + 8 * lhi) * BDIM + j] = (mv == 0) ? -1e10f : v;
                }
        }
        __syncthreads();

        // ---- softmax over j: 4 lanes per row, wave32 shfl_xor reduction ----
        {
            const int row = tid >> 2, sub = tid & 3;
            float* rp = Sim + row * BDIM + sub * 32;
            float mx = -3.0e38f;
#pragma unroll
            for (int c = 0; c < 32; ++c) mx = fmaxf(mx, rp[c]);
            mx = fmaxf(mx, __shfl_xor(mx, 1));
            mx = fmaxf(mx, __shfl_xor(mx, 2));
            float sum = 0.0f;
#pragma unroll
            for (int c = 0; c < 32; ++c) { float e = __expf(rp[c] - mx); rp[c] = e; sum += e; }
            sum += __shfl_xor(sum, 1);
            sum += __shfl_xor(sum, 2);
            const float inv = 1.0f / sum;
#pragma unroll
            for (int c = 0; c < 32; ++c) rp[c] *= inv;
        }
        __syncthreads();
        // ---- convert attn to bf16, aliasing Sim storage (register-buffered) ----
        {
            float tv[32];
#pragma unroll
            for (int kx = 0; kx < 32; ++kx) tv[kx] = Sim[tid + 256 * kx];
            __syncthreads();
#pragma unroll
            for (int kx = 0; kx < 32; ++kx) AttnB[tid + 256 * kx] = (__bf16)tv[kx];
        }

        // ---- V_h = X @ W_kv[:, INNER + h*64..]; store transposed [d][j] into KVb ----
        v8f accV[4];
#pragma unroll
        for (int nt = 0; nt < 4; ++nt) accV[nt] = zero8();
        for (int kk = 0; kk < 32; ++kk) {
            __syncthreads();
            const size_t wbase = (size_t)(kk * 32) * (2 * INNER) + INNER + h * DH;
            for (int it = tid; it < 512; it += 256) {
                int k = it >> 4, n4 = (it & 15) * 4;
                float4 v = *(const float4*)(wkv + wbase + (size_t)k * (2 * INNER) + n4);
                Wt[(n4 + 0) * WSTRIDE + k] = (__bf16)v.x;
                Wt[(n4 + 1) * WSTRIDE + k] = (__bf16)v.y;
                Wt[(n4 + 2) * WSTRIDE + k] = (__bf16)v.z;
                Wt[(n4 + 3) * WSTRIDE + k] = (__bf16)v.w;
            }
            __syncthreads();
            v16bf a = frag_ld(Xs, wv * 16, DMODEL, kk * 32, lane);
#pragma unroll
            for (int nt = 0; nt < 4; ++nt)
                accV[nt] = wmma_bf16(a, frag_ld(Wt, nt * 16, WSTRIDE, 0, lane), accV[nt]);
        }
        __syncthreads();
#pragma unroll
        for (int nt = 0; nt < 4; ++nt)
#pragma unroll
            for (int r = 0; r < 8; ++r)
                KVb[(nt * 16 + lrow) * BDIM + wv * 16 + r + 8 * lhi] = (__bf16)accV[nt][r];
        __syncthreads();

        // ---- out_h = attn(64x128) @ V(128x64); pairs share the A fragment ----
        {
            const int mt = wv >> 1, ntb = (wv & 1) * 2;
            v8f o0 = zero8(), o1 = zero8();
#pragma unroll
            for (int ks = 0; ks < 4; ++ks) {
                v16bf a = frag_ld(AttnB, mt * 16, BDIM, ks * 32, lane);
                o0 = wmma_bf16(a, frag_ld(KVb, ntb * 16, BDIM, ks * 32, lane), o0);
                o1 = wmma_bf16(a, frag_ld(KVb, (ntb + 1) * 16, BDIM, ks * 32, lane), o1);
            }
            const size_t rowbase = (size_t)wg * QN;
#pragma unroll
            for (int r = 0; r < 8; ++r) {
                const size_t ro = (rowbase + mt * 16 + r + 8 * lhi) * INNER + h * DH;
                attn_out[ro + ntb * 16 + lrow]       = o0[r];
                attn_out[ro + (ntb + 1) * 16 + lrow] = o1[r];
            }
        }
        __syncthreads();
    }
}

// =====================================================================================
// Kernel 2: LayerNorm + GEGLU FFN + residual. 32 rows / WG (512 WGs).
// FF accumulators live in registers (16 tiles x v8f per wave) across all 32 hidden
// chunks; LDS = 152 KB so two WGs can co-reside per WGP.
// =====================================================================================
__global__ __launch_bounds__(256)
void ffn_kernel(const float* __restrict__ attn_out,
                const float* __restrict__ ln_g, const float* __restrict__ ln_b,
                const float* __restrict__ W1, const float* __restrict__ b1,
                const float* __restrict__ W2, const float* __restrict__ b2,
                float* __restrict__ out)
{
    __shared__ __align__(16) __bf16 Yb[32 * INNER];      // 64 KB  [row][k]
    __shared__ __align__(16) __bf16 Hb[32 * 128];        // 8 KB   [row][k]
    __shared__ __align__(16) __bf16 Wt[1024 * WSTRIDE];  // 80 KB  transposed weight stage [n][k]

    const size_t R0 = (size_t)blockIdx.x * 32;
    const int tid = threadIdx.x, lane = tid & 31, wv = tid >> 5;
    const int lrow = lane & 15, lhi = lane >> 4;

    // ---- LayerNorm: 8 waves x 4 rows, float2 loads, wave32 shuffle reduction ----
    for (int rr = 0; rr < 4; ++rr) {
        const int row = wv * 4 + rr;
        const float2* p2 = (const float2*)(attn_out + (R0 + row) * INNER);
        float s1 = 0.0f, s2 = 0.0f;
        for (int c = lane; c < 512; c += 32) {
            float2 v = p2[c];
            s1 += v.x + v.y; s2 += v.x * v.x + v.y * v.y;
        }
#pragma unroll
        for (int off = 16; off > 0; off >>= 1) {
            s1 += __shfl_down(s1, off);
            s2 += __shfl_down(s2, off);
        }
        s1 = __shfl(s1, 0); s2 = __shfl(s2, 0);
        const float mu  = s1 * (1.0f / INNER);
        const float var = s2 * (1.0f / INNER) - mu * mu;
        const float inv = rsqrtf(var + LN_EPS);
        for (int c = lane; c < 512; c += 32) {
            float2 v = p2[c];
            store_bf2(&Yb[row * INNER + c * 2],
                      (v.x - mu) * inv * ln_g[c * 2]     + ln_b[c * 2],
                      (v.y - mu) * inv * ln_g[c * 2 + 1] + ln_b[c * 2 + 1]);
        }
    }
    __syncthreads();

    const int mtW  = wv >> 2;            // 0..1 (row tile)
    const int ntb1 = (wv & 3) * 2;       // GEMM1 n-tile pair base (of 8)
    const int ntb2 = (wv & 3) * 16;      // GEMM2 n-tile run base (of 64)

    v8f ffacc[16];
#pragma unroll
    for (int i = 0; i < 16; ++i) ffacc[i] = zero8();

    for (int ch = 0; ch < 32; ++ch) {
        // ---- GEMM1: a/g chunks = y @ W1[:, {0,4096}+ch*128 ..+128] ----
        v8f accA0, accA1, accG0, accG1;
        for (int part = 0; part < 2; ++part) {
            const int colbase = part * 4096 + ch * 128;
            v8f t0 = zero8(), t1 = zero8();
            for (int kk = 0; kk < 32; ++kk) {
                __syncthreads();
                const size_t wbase = (size_t)(kk * 32) * 8192 + colbase;
                if (kk + 1 < 32) __builtin_prefetch(W1 + wbase + (size_t)32 * 8192, 0, 3);
                for (int it = tid; it < 1024; it += 256) {     // 32k x 128n transposed stage
                    int k = it >> 5, n4 = (it & 31) * 4;
                    float4 v = *(const float4*)(W1 + wbase + (size_t)k * 8192 + n4);
                    Wt[(n4 + 0) * WSTRIDE + k] = (__bf16)v.x;
                    Wt[(n4 + 1) * WSTRIDE + k] = (__bf16)v.y;
                    Wt[(n4 + 2) * WSTRIDE + k] = (__bf16)v.z;
                    Wt[(n4 + 3) * WSTRIDE + k] = (__bf16)v.w;
                }
                __syncthreads();
                v16bf a = frag_ld(Yb, mtW * 16, INNER, kk * 32, lane);
                t0 = wmma_bf16(a, frag_ld(Wt, ntb1 * 16, WSTRIDE, 0, lane), t0);
                t1 = wmma_bf16(a, frag_ld(Wt, (ntb1 + 1) * 16, WSTRIDE, 0, lane), t1);
            }
            if (part == 0) { accA0 = t0; accA1 = t1; } else { accG0 = t0; accG1 = t1; }
        }
        // ---- GEGLU: h = (a + b1_a) * gelu_exact(g + b1_g) -> Hb ----
#pragma unroll
        for (int i = 0; i < 2; ++i) {
            const v8f aA = i ? accA1 : accA0;
            const v8f aG = i ? accG1 : accG0;
            const int ntc = (ntb1 + i) * 16;
#pragma unroll
            for (int r = 0; r < 8; ++r) {
                const int row = mtW * 16 + r + 8 * lhi;
                const int col = ntc + lrow;
                float av = aA[r] + b1[ch * 128 + col];
                float gv = aG[r] + b1[4096 + ch * 128 + col];
                float ge = 0.5f * gv * (1.0f + erff(gv * 0.70710678118f));
                Hb[row * 128 + col] = (__bf16)(av * ge);
            }
        }
        // ---- GEMM2: ffacc += H(32x128) @ W2[ch*128.., :], one A per k-step ----
        for (int ks = 0; ks < 4; ++ks) {
            __syncthreads();                                  // Wt free + Hb visible
            const size_t wbase = (size_t)(ch * 128 + ks * 32) * INNER;
            for (int it = tid; it < 8192; it += 256) {        // 32k x 1024n transposed stage
                int k = it >> 8, n4 = (it & 255) * 4;
                float4 v = *(const float4*)(W2 + wbase + (size_t)k * INNER + n4);
                Wt[(n4 + 0) * WSTRIDE + k] = (__bf16)v.x;
                Wt[(n4 + 1) * WSTRIDE + k] = (__bf16)v.y;
                Wt[(n4 + 2) * WSTRIDE + k] = (__bf16)v.z;
                Wt[(n4 + 3) * WSTRIDE + k] = (__bf16)v.w;
            }
            __syncthreads();
            v16bf a = frag_ld(Hb, mtW * 16, 128, ks * 32, lane);
#pragma unroll
            for (int i = 0; i < 16; ++i)
                ffacc[i] = wmma_bf16(a, frag_ld(Wt, (ntb2 + i) * 16, WSTRIDE, 0, lane), ffacc[i]);
        }
    }

    // ---- epilogue: out = ff + b2 + y (from registers) ----
#pragma unroll
    for (int i = 0; i < 16; ++i) {
        const int ntc = (ntb2 + i) * 16;
#pragma unroll
        for (int r = 0; r < 8; ++r) {
            const int row = mtW * 16 + r + 8 * lhi;
            const int col = ntc + lrow;
            out[(R0 + row) * INNER + col] = ffacc[i][r] + b2[col] + (float)Yb[row * INNER + col];
        }
    }
}

extern "C" void kernel_launch(void* const* d_in, const int* in_sizes, int n_in,
                              void* d_out, int out_size, void* d_ws, size_t ws_size,
                              hipStream_t stream) {
    const float* x    = (const float*)d_in[0];
    const int*   mask = (const int*)d_in[1];
    const float* q    = (const float*)d_in[2];
    const float* wkv  = (const float*)d_in[3];
    const float* ln_g = (const float*)d_in[4];
    const float* ln_b = (const float*)d_in[5];
    const float* W1   = (const float*)d_in[6];
    const float* b1   = (const float*)d_in[7];
    const float* W2   = (const float*)d_in[8];
    const float* b2   = (const float*)d_in[9];
    float* outp     = (float*)d_out;
    float* attn_out = (float*)d_ws;   // 16384 x 1024 f32 = 64 MB scratch

    attn_kernel<<<SDIM * NDIM, 256, 0, stream>>>(x, mask, q, wkv, attn_out);
    ffn_kernel<<<(SDIM * NDIM * QN) / 32, 256, 0, stream>>>(attn_out, ln_g, ln_b,
                                                            W1, b1, W2, b2, outp);
}
